// attentiona_9010841387745
// MI455X (gfx1250) — matrix operand order repaired and checked by reading, not compiled, not run on hardware
//
#include <hip/hip_runtime.h>
#include <hip/hip_bf16.h>

// Problem constants
#define BB   2
#define SS   2048
#define DD   512
#define HH   8
#define DHD  64
#define BSR  (BB * SS)            // 4096 token rows
#define TOT  (BB * HH * SS * DHD) // 2097152 elements in (b,h,s,d) tensors

typedef __attribute__((ext_vector_type(16))) __bf16 v16bf;
typedef __attribute__((ext_vector_type(8)))  float  v8f;

struct U32x8 { uint4 lo, hi; };

static __device__ __forceinline__ v16bf frag_from(uint4 lo, uint4 hi) {
  U32x8 t; t.lo = lo; t.hi = hi;
  return __builtin_bit_cast(v16bf, t);
}

// A fragment (16x32 bf16, row-major src): lane<16 -> row=lane, K in [0,8)+[16,24)
//                                          lane>=16 -> row=lane-16, K in [8,16)+[24,32)
static __device__ __forceinline__ v16bf load_a_frag(const __bf16* base, int ld, int lane) {
  int row = lane & 15;
  int kb  = (lane >> 4) * 8;
  const __bf16* p = base + row * ld + kb;
  uint4 lo = *(const uint4*)(p);
  uint4 hi = *(const uint4*)(p + 16);
  return frag_from(lo, hi);
}

// B fragment (32x16 bf16) from an N-major (n rows, K contiguous) source:
//   lane<16 -> col=lane, K [0,8)+[16,24);  lane>=16 -> col=lane-16, K [8,16)+[24,32)
static __device__ __forceinline__ v16bf load_b_frag(const __bf16* baseNxK, int ld, int lane) {
  int col = lane & 15;
  int kb  = (lane >> 4) * 8;
  const uint4* p = (const uint4*)(baseNxK + col * ld + kb);
  return frag_from(p[0], p[2]);
}

static __device__ __forceinline__ v8f wmma_bf16(v16bf a, v16bf b, v8f c) {
  return __builtin_amdgcn_wmma_f32_16x16x32_bf16(false, a, false, b, (short)0, c, false, false);
}

static __device__ __forceinline__ __bf16 f2bf(float f) {
  unsigned u = __builtin_bit_cast(unsigned, f);
  u += 0x7fffu + ((u >> 16) & 1u);   // round-to-nearest-even
  unsigned short h = (unsigned short)(u >> 16);
  return __builtin_bit_cast(__bf16, h);
}

// Async global->LDS 16-byte copy (gfx1250): VDST = LDS byte offset (low 32 bits of
// the generic shared-memory address per the LDS aperture truncation rule).
static __device__ __forceinline__ void async_cp16(void* lds_dst, const void* gsrc) {
  unsigned           loff = (unsigned)(size_t)lds_dst;
  unsigned long long ga   = (unsigned long long)(size_t)gsrc;
  asm volatile("global_load_async_to_lds_b128 %0, %1, off"
               :: "v"(loff), "v"(ga) : "memory");
}

// ---------------------------------------------------------------- utilities
__global__ void fill0_kernel(float* p, int n) {
  for (int i = blockIdx.x * blockDim.x + threadIdx.x; i < n; i += gridDim.x * blockDim.x)
    p[i] = 0.0f;
}

__global__ void cvt_bf_kernel(const float* __restrict__ src, __bf16* __restrict__ dst, int n) {
  for (int i = blockIdx.x * blockDim.x + threadIdx.x; i < n; i += gridDim.x * blockDim.x)
    dst[i] = f2bf(src[i]);
}

// ---------------------------------------------------------------- LayerNorm over D=512 rows -> bf16
__global__ void __launch_bounds__(256)
ln_rows_kernel(const float* __restrict__ x, const float* __restrict__ w,
               __bf16* __restrict__ out) {
  __shared__ float red[8];
  const int row = blockIdx.x;
  const float* xr = x + (size_t)row * DD;
  const int t = threadIdx.x, lane = t & 31, wv = t >> 5;
  float a0 = xr[t], a1 = xr[t + 256];
  float s = a0 + a1;
  #pragma unroll
  for (int mk = 16; mk; mk >>= 1) s += __shfl_xor(s, mk, 32);
  if (lane == 0) red[wv] = s;
  __syncthreads();
  float mean = 0.f;
  #pragma unroll
  for (int i = 0; i < 8; ++i) mean += red[i];
  mean *= (1.0f / DD);
  float d0 = a0 - mean, d1 = a1 - mean;
  float q = d0 * d0 + d1 * d1;
  #pragma unroll
  for (int mk = 16; mk; mk >>= 1) q += __shfl_xor(q, mk, 32);
  __syncthreads();
  if (lane == 0) red[wv] = q;
  __syncthreads();
  float var = 0.f;
  #pragma unroll
  for (int i = 0; i < 8; ++i) var += red[i];
  var *= (1.0f / DD);
  float r = rsqrtf(var + 1e-5f);
  out[(size_t)row * DD + t]       = f2bf(d0 * r * w[t]);
  out[(size_t)row * DD + t + 256] = f2bf(d1 * r * w[t + 256]);
}

// ---------------------------------------------------------------- generic bf16 WMMA GEMM
// out = (A @ Wt^T + bias) * scale ; A: MxK row-major, Wt: NxK row-major; M,N multiples of 128.
// 256 threads = 8 waves; block tile 128x128; wave tile 32x64; async double-buffered staging.
// mode 0: out row-major MxN.  mode 1: out in (b,h,s,d) head layout (M=b*S+s, N=h*64+d).
__global__ void __launch_bounds__(256)
gemm_bf16_kernel(const __bf16* __restrict__ A, const __bf16* __restrict__ Wt,
                 const float* __restrict__ bias, float scale,
                 float* __restrict__ out, int M, int N, int K, int mode) {
  __shared__ __bf16 Ash[2][128 * 40];
  __shared__ __bf16 Bsh[2][128 * 40];
  const int tid = threadIdx.x, lane = tid & 31, wv = tid >> 5;
  const int wm = wv & 3, wn = wv >> 2;                 // 4x2 wave grid
  const int n0 = blockIdx.x * 128, m0 = blockIdx.y * 128;
  const __bf16* Ag = A  + (size_t)m0 * K;
  const __bf16* Bg = Wt + (size_t)n0 * K;

  // per-thread chunk assignment for a 128x32 tile = 512 x 16B chunks, 4 per thread
  const int r0 = tid >> 1,        c0 = (tid & 1) * 16;        // chunks 0..511
  const int r1 = (tid + 256) >> 1, c1 = c0;                    // chunks 512..1023 (rows 128..255)
  auto stage = [&](const __bf16* g, __bf16* l, int k0) {
    async_cp16(l + r0 * 40 + c0,      g + (size_t)r0 * K + k0 + c0);
    async_cp16(l + r0 * 40 + c0 + 8,  g + (size_t)r0 * K + k0 + c0 + 8);
    async_cp16(l + r1 * 40 + c1,      g + (size_t)r1 * K + k0 + c1);
    async_cp16(l + r1 * 40 + c1 + 8,  g + (size_t)r1 * K + k0 + c1 + 8);
  };

  v8f acc[2][4] = {};
  const int nk = K / 32;
  stage(Ag, Ash[0], 0);
  stage(Bg, Bsh[0], 0);
  for (int ks = 0; ks < nk; ++ks) {
    const int cur = ks & 1;
    if (ks + 1 < nk) {
      stage(Ag, Ash[cur ^ 1], (ks + 1) * 32);
      stage(Bg, Bsh[cur ^ 1], (ks + 1) * 32);
      asm volatile("s_wait_asynccnt 0x8" ::: "memory");  // prev stage (8 instrs) landed
    } else {
      asm volatile("s_wait_asynccnt 0x0" ::: "memory");
    }
    __syncthreads();
    v16bf a0 = load_a_frag(&Ash[cur][(wm * 32 + 0)  * 40], 40, lane);
    v16bf a1 = load_a_frag(&Ash[cur][(wm * 32 + 16) * 40], 40, lane);
    #pragma unroll
    for (int nf = 0; nf < 4; ++nf) {
      v16bf b = load_b_frag(&Bsh[cur][(wn * 64 + nf * 16) * 40], 40, lane);
      acc[0][nf] = wmma_bf16(a0, b, acc[0][nf]);
      acc[1][nf] = wmma_bf16(a1, b, acc[1][nf]);
    }
    __syncthreads();
  }
  const int nl = lane & 15, mhi = (lane >> 4) * 8;
  #pragma unroll
  for (int am = 0; am < 2; ++am) {
    #pragma unroll
    for (int nf = 0; nf < 4; ++nf) {
      #pragma unroll
      for (int r = 0; r < 8; ++r) {
        int m = m0 + wm * 32 + am * 16 + mhi + r;
        int n = n0 + wn * 64 + nf * 16 + nl;
        float v = acc[am][nf][r];
        if (bias) v += bias[n];
        v *= scale;
        if (mode == 0) {
          out[(size_t)m * N + n] = v;
        } else {
          int b = m >> 11, s = m & (SS - 1), h = n >> 6, d = n & 63;
          out[(((size_t)(b * HH + h)) * SS + s) * DHD + d] = v;
        }
      }
    }
  }
}

// ---------------------------------------------------------------- per-head linear + RoPE + LN
// mode 0 (q): hat = bf16(LN(rope(src @ wt^T + bias)))
// mode 1 (k): raw = rope(...), hat = bf16(LN(raw))
// mode 2 (v): raw = src @ wt^T + bias, vT = bf16(raw) transposed to (b,h,d,s)
__global__ void __launch_bounds__(128)
head_linear_kernel(const float* __restrict__ src, const __bf16* __restrict__ wt,
                   const float* __restrict__ bias, const float* __restrict__ lnb_w,
                   const float* __restrict__ theta_p,
                   __bf16* __restrict__ hat, float* __restrict__ raw,
                   __bf16* __restrict__ vT, int mode) {
  __shared__ __bf16 Ash[64 * 72];
  __shared__ float  Csh[64 * 72];
  const int tid = threadIdx.x, lane = tid & 31, wv = tid >> 5;
  const int bh = blockIdx.y, s0 = blockIdx.x * 64;
  const size_t rowbase = (size_t)bh * SS + s0;
  #pragma unroll
  for (int j = 0; j < 32; ++j) {
    int i = tid + j * 128;
    int r = i >> 6, c = i & 63;
    Ash[r * 72 + c] = f2bf(src[(rowbase + r) * DHD + c]);
  }
  __syncthreads();
  v8f acc[4] = {};
  #pragma unroll
  for (int kf = 0; kf < 2; ++kf) {
    v16bf a = load_a_frag(&Ash[(wv * 16) * 72 + kf * 32], 72, lane);
    #pragma unroll
    for (int nf = 0; nf < 4; ++nf) {
      v16bf b = load_b_frag(wt + (nf * 16) * DHD + kf * 32, DHD, lane);
      acc[nf] = wmma_bf16(a, b, acc[nf]);
    }
  }
  const int nl = lane & 15, mhi = (lane >> 4) * 8;
  #pragma unroll
  for (int nf = 0; nf < 4; ++nf)
    #pragma unroll
    for (int r = 0; r < 8; ++r) {
      int n = nf * 16 + nl;
      Csh[(wv * 16 + mhi + r) * 72 + n] = acc[nf][r] + bias[n];
    }
  __syncthreads();
  if (tid < 64) {
    const int sg = s0 + tid;
    float* row = &Csh[tid * 72];
    if (mode < 2) {
      const float th = *theta_p;
      const float hi_c = 3431.1590844f; // 2595 * log10(1 + 4000/200)
      for (int i = 0; i < 32; ++i) {
        float e    = ((hi_c / 31.0f) * (float)i) * (1.0f / 2595.0f);
        float mel  = __expf(e * 2.3025850929940457f) - 1.0f;
        float freq = (th / 220.0f) * (0.2f * mel);
        float ang  = (float)sg * freq;
        float cs, sn;
        __sincosf(ang, &sn, &cs);
        float a = row[2 * i], b = row[2 * i + 1];
        row[2 * i]     = a * cs - b * sn;
        row[2 * i + 1] = a * sn + b * cs;
      }
      if (mode == 1) {
        for (int d = 0; d < 64; ++d) raw[(rowbase + tid) * DHD + d] = row[d];
      }
      float mu = 0.f;
      for (int d = 0; d < 64; ++d) mu += row[d];
      mu *= (1.0f / 64.0f);
      float var = 0.f;
      for (int d = 0; d < 64; ++d) { float t2 = row[d] - mu; var += t2 * t2; }
      var *= (1.0f / 64.0f);
      float rs = rsqrtf(var + 1e-5f);
      for (int d = 0; d < 64; ++d)
        hat[(rowbase + tid) * DHD + d] = f2bf((row[d] - mu) * rs * lnb_w[d]);
    } else {
      for (int d = 0; d < 64; ++d) {
        float v = row[d];
        raw[(rowbase + tid) * DHD + d] = v;
        vT[((size_t)bh * DHD + d) * SS + s0 + tid] = f2bf(v);
      }
    }
  }
}

// ---------------------------------------------------------------- flash attention (per b,h, 64 q rows)
__global__ void __launch_bounds__(128)
flash_attn_kernel(const __bf16* __restrict__ qhat, const __bf16* __restrict__ khat,
                  const __bf16* __restrict__ vT, const float* __restrict__ prev,
                  float* __restrict__ iter_out, float* __restrict__ diffsum) {
  __shared__ __bf16 Psh[64 * 72];
  const int tid = threadIdx.x, lane = tid & 31, wv = tid >> 5;
  const int bh = blockIdx.y, b = bh >> 3, h = bh & 7;
  const int q0 = blockIdx.x * 64 + wv * 16;
  const size_t qbase = ((size_t)bh * SS + q0) * DHD;
  const v16bf Aq0 = load_a_frag(qhat + qbase + 0,  DHD, lane);
  const v16bf Aq1 = load_a_frag(qhat + qbase + 32, DHD, lane);
  v8f acc[4] = {};
  float m_run[8], l_run[8];
  #pragma unroll
  for (int r = 0; r < 8; ++r) { m_run[r] = -1e30f; l_run[r] = 0.f; }
  __bf16* Pw = &Psh[wv * 16 * 72];
  const int nl = lane & 15, mhi = (lane >> 4) * 8;
  for (int kt = 0; kt < SS / 64; ++kt) {
    if (kt + 1 < SS / 64) {  // pull next key tile toward the caches
      __builtin_prefetch(khat + ((size_t)bh * SS + (kt + 1) * 64) * DHD, 0, 0);
      __builtin_prefetch(vT + ((size_t)bh * DHD) * SS + (kt + 1) * 64, 0, 0);
    }
    // --- scores: issue all fragment loads first, then the wmma chain
    v16bf kbf[4][2];
    #pragma unroll
    for (int nf = 0; nf < 4; ++nf) {
      const __bf16* kb = khat + ((size_t)bh * SS + kt * 64 + nf * 16) * DHD;
      kbf[nf][0] = load_b_frag(kb + 0,  DHD, lane);
      kbf[nf][1] = load_b_frag(kb + 32, DHD, lane);
    }
    v8f sc[4];
    #pragma unroll
    for (int nf = 0; nf < 4; ++nf) {
      v8f c = {};
      c = wmma_bf16(Aq0, kbf[nf][0], c);
      c = wmma_bf16(Aq1, kbf[nf][1], c);
      sc[nf] = c * 0.125f; // 1/sqrt(64)
    }
    float newm[8], alpha[8], rsum[8];
    #pragma unroll
    for (int r = 0; r < 8; ++r) {
      float t = fmaxf(fmaxf(sc[0][r], sc[1][r]), fmaxf(sc[2][r], sc[3][r]));
      #pragma unroll
      for (int mk = 1; mk < 16; mk <<= 1) t = fmaxf(t, __shfl_xor(t, mk, 32));
      newm[r]  = fmaxf(m_run[r], t);
      alpha[r] = __expf(m_run[r] - newm[r]);
      m_run[r] = newm[r];
      rsum[r]  = 0.f;
    }
    #pragma unroll
    for (int nf = 0; nf < 4; ++nf)
      #pragma unroll
      for (int r = 0; r < 8; ++r) {
        float p = __expf(sc[nf][r] - newm[r]);
        rsum[r] += p;
        Pw[(mhi + r) * 72 + nf * 16 + nl] = f2bf(p);
      }
    #pragma unroll
    for (int r = 0; r < 8; ++r) {
      float t = rsum[r];
      #pragma unroll
      for (int mk = 1; mk < 16; mk <<= 1) t += __shfl_xor(t, mk, 32);
      l_run[r] = l_run[r] * alpha[r] + t;
    }
    #pragma unroll
    for (int nf = 0; nf < 4; ++nf)
      #pragma unroll
      for (int r = 0; r < 8; ++r) acc[nf][r] *= alpha[r];
    asm volatile("s_wait_dscnt 0" ::: "memory"); // P stores -> same-wave ds reads
    // --- P @ V: load all fragments, then wmma chain
    v16bf Pa[2], vbf[2][4];
    #pragma unroll
    for (int kf = 0; kf < 2; ++kf) {
      Pa[kf] = load_a_frag(Pw + kf * 32, 72, lane);
      #pragma unroll
      for (int nf = 0; nf < 4; ++nf)
        vbf[kf][nf] = load_b_frag(vT + ((size_t)bh * DHD + nf * 16) * SS + kt * 64 + kf * 32,
                                  SS, lane);
    }
    #pragma unroll
    for (int kf = 0; kf < 2; ++kf)
      #pragma unroll
      for (int nf = 0; nf < 4; ++nf)
        acc[nf] = wmma_bf16(Pa[kf], vbf[kf][nf], acc[nf]);
  }
  float ldiff = 0.f;
  #pragma unroll
  for (int nf = 0; nf < 4; ++nf)
    #pragma unroll
    for (int r = 0; r < 8; ++r) {
      float o  = acc[nf][r] / l_run[r];
      int s_a  = q0 + mhi + r;
      int d_a  = nf * 16 + nl;
      // reference's reshape scramble: (h_a,s_a,d_a) -> (h2,s2,d2)
      int h2 = s_a >> 8;
      int s2 = (s_a * 8 + h) & (SS - 1);
      size_t addr = (((size_t)(b * HH + h2)) * SS + s2) * DHD + d_a;
      ldiff += fabsf(o - prev[addr]);
      iter_out[addr] = o;
    }
  #pragma unroll
  for (int mk = 1; mk < 32; mk <<= 1) ldiff += __shfl_xor(ldiff, mk, 32);
  if (lane == 0) atomicAdd(diffsum, ldiff);
}

// ---------------------------------------------------------------- convergence control (1 thread)
__global__ void control_kernel(const float* __restrict__ diffsum, int* __restrict__ done,
                               int* __restrict__ flags, int it) {
  if (threadIdx.x == 0 && blockIdx.x == 0) {
    float diff = diffsum[it] * (1.0f / (float)TOT);
    int brk    = (diff < 0.01f + 0.1f * diff) && (it > 0);
    int active = (*done == 0);
    flags[it * 2 + 0] = active;
    flags[it * 2 + 1] = active && !brk;
    *done = (*done) || brk;
  }
}

__global__ void apply_update_kernel(const float* __restrict__ iter_out,
                                    float* __restrict__ attn, float* __restrict__ prevb,
                                    float* __restrict__ q_cur, float* __restrict__ kb,
                                    float* __restrict__ vb,
                                    const float* __restrict__ kr, const float* __restrict__ vl,
                                    const int* __restrict__ flags, int it, int n) {
  const int active = flags[it * 2 + 0];
  const int upd    = flags[it * 2 + 1];
  for (int i = blockIdx.x * blockDim.x + threadIdx.x; i < n; i += gridDim.x * blockDim.x) {
    float io = iter_out[i];
    if (active) attn[i] = io;
    if (upd) {
      prevb[i] = io;
      q_cur[i] += io;
      kb[i] = kr[i];
      vb[i] = vl[i];
    }
  }
}

// (b,h,s,d) f32 -> (b*s, h*64+d) bf16 for the output projection
__global__ void pack_attn_kernel(const float* __restrict__ attn, __bf16* __restrict__ a3, int n) {
  for (int i = blockIdx.x * blockDim.x + threadIdx.x; i < n; i += gridDim.x * blockDim.x) {
    int d = i & 63, s = (i >> 6) & (SS - 1), h = (i >> 17) & 7, b = i >> 20;
    a3[((size_t)(b * SS + s)) * DD + h * DHD + d] = f2bf(attn[i]);
  }
}

// ---------------------------------------------------------------- host launch
extern "C" void kernel_launch(void* const* d_in, const int* in_sizes, int n_in,
                              void* d_out, int out_size, void* d_ws, size_t ws_size,
                              hipStream_t stream) {
  const float* x     = (const float*)d_in[0];
  const float* qw    = (const float*)d_in[1];
  const float* qb    = (const float*)d_in[2];
  const float* kw    = (const float*)d_in[3];
  const float* vw    = (const float*)d_in[4];
  const float* vb    = (const float*)d_in[5];
  const float* ow    = (const float*)d_in[6];
  const float* ob    = (const float*)d_in[7];
  const float* lna_w = (const float*)d_in[8];
  const float* lnb_w = (const float*)d_in[9];
  const float* lqw   = (const float*)d_in[10];
  const float* lqb   = (const float*)d_in[11];
  const float* lkw   = (const float*)d_in[12];
  const float* lkb   = (const float*)d_in[13];
  const float* lvw   = (const float*)d_in[14];
  const float* lvb   = (const float*)d_in[15];
  const float* theta = (const float*)d_in[16];
  float* out = (float*)d_out;

  char* wsp = (char*)d_ws;
  auto alloc = [&](size_t bytes) -> void* {
    void* p = (void*)wsp;
    wsp += (bytes + 255) & ~(size_t)255;
    return p;
  };
  __bf16* xn_bf  = (__bf16*)alloc((size_t)BSR * DD * 2);
  __bf16* qw_bf  = (__bf16*)alloc((size_t)DD * DD * 2);
  __bf16* kw_bf  = (__bf16*)alloc((size_t)DD * DD * 2);
  __bf16* vw_bf  = (__bf16*)alloc((size_t)DD * DD * 2);
  __bf16* ow_bf  = (__bf16*)alloc((size_t)DD * DD * 2);
  __bf16* lqw_bf = (__bf16*)alloc((size_t)DHD * DHD * 2);
  __bf16* lkw_bf = (__bf16*)alloc((size_t)DHD * DHD * 2);
  __bf16* lvw_bf = (__bf16*)alloc((size_t)DHD * DHD * 2);
  float*  q_cur  = (float*)alloc((size_t)TOT * 4);
  float*  kbuf   = (float*)alloc((size_t)TOT * 4);
  float*  vbuf   = (float*)alloc((size_t)TOT * 4);
  float*  prevb  = (float*)alloc((size_t)TOT * 4);
  float*  attn   = (float*)alloc((size_t)TOT * 4);
  float*  iter   = (float*)alloc((size_t)TOT * 4);
  float*  kr     = (float*)alloc((size_t)TOT * 4);
  float*  vl     = (float*)alloc((size_t)TOT * 4);
  __bf16* qhat   = (__bf16*)alloc((size_t)TOT * 2);
  __bf16* khat   = (__bf16*)alloc((size_t)TOT * 2);
  __bf16* vhatT  = (__bf16*)alloc((size_t)TOT * 2);
  __bf16* a3     = (__bf16*)alloc((size_t)BSR * DD * 2);
  float*  scal   = (float*)alloc(64 * 4); // [0..2]=diffsum, [3]=done, [4..9]=flags
  float*  diffsum = scal;
  int*    done    = (int*)(scal + 3);
  int*    flags   = (int*)(scal + 4);

  const float qkscale = 0.35355339059327373f; // 64^-0.25

  fill0_kernel<<<1, 64, 0, stream>>>(scal, 16);
  fill0_kernel<<<1024, 256, 0, stream>>>(prevb, TOT);

  ln_rows_kernel<<<BSR, 256, 0, stream>>>(x, lna_w, xn_bf);

  cvt_bf_kernel<<<512, 256, 0, stream>>>(qw,  qw_bf,  DD * DD);
  cvt_bf_kernel<<<512, 256, 0, stream>>>(kw,  kw_bf,  DD * DD);
  cvt_bf_kernel<<<512, 256, 0, stream>>>(vw,  vw_bf,  DD * DD);
  cvt_bf_kernel<<<512, 256, 0, stream>>>(ow,  ow_bf,  DD * DD);
  cvt_bf_kernel<<<16, 256, 0, stream>>>(lqw, lqw_bf, DHD * DHD);
  cvt_bf_kernel<<<16, 256, 0, stream>>>(lkw, lkw_bf, DHD * DHD);
  cvt_bf_kernel<<<16, 256, 0, stream>>>(lvw, lvw_bf, DHD * DHD);

  dim3 gqkv(DD / 128, BSR / 128);   // 128x128 block tiles
  gemm_bf16_kernel<<<gqkv, 256, 0, stream>>>(xn_bf, qw_bf, qb,      qkscale, q_cur, BSR, DD, DD, 1);
  gemm_bf16_kernel<<<gqkv, 256, 0, stream>>>(xn_bf, kw_bf, nullptr, qkscale, kbuf,  BSR, DD, DD, 1);
  gemm_bf16_kernel<<<gqkv, 256, 0, stream>>>(xn_bf, vw_bf, vb,      1.0f,    vbuf,  BSR, DD, DD, 1);

  dim3 ghead(SS / 64, BB * HH);
  for (int it = 0; it < 3; ++it) {
    head_linear_kernel<<<ghead, 128, 0, stream>>>(q_cur, lqw_bf, lqb, lnb_w, theta,
                                                  qhat, nullptr, nullptr, 0);
    head_linear_kernel<<<ghead, 128, 0, stream>>>(kbuf, lkw_bf, lkb, lnb_w, theta,
                                                  khat, kr, nullptr, 1);
    head_linear_kernel<<<ghead, 128, 0, stream>>>(vbuf, lvw_bf, lvb, lnb_w, theta,
                                                  nullptr, vl, vhatT, 2);
    flash_attn_kernel<<<ghead, 128, 0, stream>>>(qhat, khat, vhatT, prevb, iter,
                                                 diffsum + it);
    control_kernel<<<1, 32, 0, stream>>>(diffsum, done, flags, it);
    apply_update_kernel<<<1024, 256, 0, stream>>>(iter, attn, prevb, q_cur, kbuf, vbuf,
                                                  kr, vl, flags, it, TOT);
  }

  pack_attn_kernel<<<1024, 256, 0, stream>>>(attn, a3, TOT);
  gemm_bf16_kernel<<<gqkv, 256, 0, stream>>>(a3, ow_bf, ob, 1.0f, out, BSR, DD, DD, 0);
}
